// ThinkingLearnableDelayRNN_21440476741884
// MI455X (gfx1250) — matrix operand order repaired and checked by reading, not compiled, not run on hardware
//
#include <hip/hip_runtime.h>
#include <cstdint>
#include <cstddef>

// Problem constants (fixed in the reference file)
#define TB   64      // batch
#define TT   128     // time steps
#define THH  256     // hidden
#define TIN  256     // input dim
#define TV   256     // vocab
#define TD   33      // delay slots
#define TN   125     // decode steps (T-3)
#define DHSZ (TD*THH)   // per-sample decode buffer floats (8448)

typedef __attribute__((ext_vector_type(16))) _Float16 v16h;
typedef __attribute__((ext_vector_type(8)))  _Float16 v8h;
typedef __attribute__((ext_vector_type(8)))  float    v8f;
typedef __attribute__((ext_vector_type(4)))  float    v4f;

// ---------------- WMMA helpers (gfx1250, wave32) ----------------

__device__ __forceinline__ v8f wmma_acc(v16h a, v16h b, v8f c) {
    // D = A(16x32 f16) * B(32x16 f16) + C(16x16 f32)
    return __builtin_amdgcn_wmma_f32_16x16x32_f16(false, a, false, b, (short)0, c,
                                                  false, false);
}

// A fragment: lane l -> row (l&15); K = kb + 8*(l>>4) + {0..7} and +{16..23}
__device__ __forceinline__ v16h frag_a_f16(const _Float16* base, int ld, int kb) {
    int lane = threadIdx.x & 31;
    const _Float16* p = base + (size_t)(lane & 15) * ld + kb + 8 * (lane >> 4);
    v8h lo = *(const v8h*)(p);
    v8h hi = *(const v8h*)(p + 16);
    v16h f;
#pragma unroll
    for (int e = 0; e < 8; ++e) { f[e] = lo[e]; f[e + 8] = hi[e]; }
    return f;
}

__device__ __forceinline__ v16h frag_a_f32(const float* base, int ld, int kb) {
    int lane = threadIdx.x & 31;
    const float* p = base + (size_t)(lane & 15) * ld + kb + 8 * (lane >> 4);
    const v4f* p4 = (const v4f*)p;
    v4f a0 = p4[0], a1 = p4[1], a2 = p4[4], a3 = p4[5];
    v16h f;
#pragma unroll
    for (int e = 0; e < 4; ++e) {
        f[e]      = (_Float16)a0[e];
        f[e + 4]  = (_Float16)a1[e];
        f[e + 8]  = (_Float16)a2[e];
        f[e + 12] = (_Float16)a3[e];
    }
    return f;
}

// A fragment gathered per-row from the decode delay buffer at each row's ptr
__device__ __forceinline__ v16h frag_a_gather(const float* dbuf, const int* ptr,
                                              int m0, int kb) {
    int lane = threadIdx.x & 31;
    int m = m0 + (lane & 15);
    const float* p = dbuf + (size_t)m * DHSZ + (size_t)ptr[m] * THH + kb + 8 * (lane >> 4);
    const v4f* p4 = (const v4f*)p;
    v4f a0 = p4[0], a1 = p4[1], a2 = p4[4], a3 = p4[5];
    v16h f;
#pragma unroll
    for (int e = 0; e < 4; ++e) {
        f[e]      = (_Float16)a0[e];
        f[e + 4]  = (_Float16)a1[e];
        f[e + 8]  = (_Float16)a2[e];
        f[e + 12] = (_Float16)a3[e];
    }
    return f;
}

// B fragment: lane l -> col (l&15); K = kb + 8*(l>>4) + {0..7} and +{16..23}. Source is
// W row-major (N x K) so B[k][n] = W[n][k] is contiguous in k.
__device__ __forceinline__ v16h frag_b_f16(const _Float16* base, int ld, int kb) {
    int lane = threadIdx.x & 31;
    const _Float16* p = base + (size_t)(lane & 15) * ld + kb + 8 * (lane >> 4);
    v8h lo = *(const v8h*)(p);
    v8h hi = *(const v8h*)(p + 16);
    v16h f;
#pragma unroll
    for (int e = 0; e < 8; ++e) { f[e] = lo[e]; f[e + 8] = hi[e]; }
    return f;
}

// splitmix64-derived Gumbel noise (threefry not reproducible; structural match)
__device__ __forceinline__ float gumbelf(uint32_t b, uint32_t k, uint32_t c) {
    uint64_t z = (((uint64_t)b << 40) ^ ((uint64_t)k << 20) ^ (uint64_t)c) +
                 0x9E3779B97F4A7C15ull;
    z = (z ^ (z >> 30)) * 0xBF58476D1CE4E5B9ull;
    z = (z ^ (z >> 27)) * 0x94D049BB133111EBull;
    z ^= z >> 31;
    float u = ((float)(uint32_t)(z >> 40) + 0.5f) * (1.0f / 16777216.0f);
    return -__logf(-__logf(u));
}

// ---------------- prep kernels ----------------

__global__ void k_cvt_f16(const float* __restrict__ src, _Float16* __restrict__ dst, int n) {
    int i = blockIdx.x * blockDim.x + threadIdx.x;
    if (i < n) dst[i] = (_Float16)src[i];
}

__global__ void k_zero_f32(float* __restrict__ p, int n) {
    int i = blockIdx.x * blockDim.x + threadIdx.x;
    if (i < n) p[i] = 0.0f;
}

// cw[d,h,i] = relu(-|(d - clip(tau,1,32)) * inv^2| + inv) * lateral[h,i], f16
__global__ void k_prep_cw(const float* __restrict__ tau, const float* __restrict__ lateral,
                          _Float16* __restrict__ cw16) {
    int idx = blockIdx.x * blockDim.x + threadIdx.x;  // < TD*THH*TIN
    int d = idx / (THH * TIN);
    int hi = idx - d * (THH * TIN);
    float tc = fminf(fmaxf(tau[hi], 1.0f), 32.0f);
    const float inv = 1.0f / 16.0f;
    float c = -fabsf(((float)d - tc) * (inv * inv)) + inv;
    c = fmaxf(c, 0.0f);
    cw16[idx] = (_Float16)(c * lateral[hi]);
}

__global__ void k_init_dec(int* __restrict__ done, int* __restrict__ steps) {
    int b = threadIdx.x;
    if (b < TB) { done[b] = 0; steps[b] = 0; }
}

// cur16[b,:] = f16(x[b, lengths[b]-1, :])
__global__ void k_gather_decin(const float* __restrict__ x, const int* __restrict__ lengths,
                               _Float16* __restrict__ cur16) {
    int b = blockIdx.x, i = threadIdx.x;
    int t = lengths[b] - 1;
    cur16[b * TIN + i] = (_Float16)x[((size_t)b * TT + t) * TIN + i];
}

// ---------------- encoder (one step per launch trio) ----------------

// h16 = f16(tanh(x_t @ w_aff^T + b_aff + buf[ptr]))   (64x256)
__global__ void k_enc_h(const float* __restrict__ x, int t,
                        const _Float16* __restrict__ w_aff16, const float* __restrict__ b_aff,
                        const float* __restrict__ buf, int ptr,
                        _Float16* __restrict__ h16) {
    int wid = (blockIdx.x * blockDim.x + threadIdx.x) >> 5;  // 0..63
    int m0 = (wid >> 4) << 4;
    int n0 = (wid & 15) << 4;
    const float* A = x + (size_t)m0 * (TT * TIN) + (size_t)t * TIN;
    const _Float16* Bm = w_aff16 + (size_t)n0 * TIN;
    v8f acc = {};
#pragma unroll
    for (int kc = 0; kc < TIN; kc += 32)
        acc = wmma_acc(frag_a_f32(A, TT * TIN, kc), frag_b_f16(Bm, TIN, kc), acc);
    int lane = threadIdx.x & 31;
    int n = n0 + (lane & 15);
    const float* bp = buf + (size_t)ptr * TB * THH;
#pragma unroll
    for (int r = 0; r < 8; ++r) {
        int m = m0 + r + 8 * (lane >> 4);
        float hv = tanhf(acc[r] + b_aff[n] + bp[m * THH + n]);
        h16[m * THH + n] = (_Float16)hv;
    }
}

// buf[(ptr+d)%33] += h @ cw[d]^T for d=1..32 ; buf[ptr] = 0
// One wave computes a full 64x16 column strip: B fragment register-reused
// across 4 M-tiles (32 WMMAs/wave, minimal cw traffic from L2).
__global__ void k_enc_update(const _Float16* __restrict__ h16,
                             const _Float16* __restrict__ cw16,
                             float* __restrict__ buf, int ptr) {
    int wid = (blockIdx.x * blockDim.x + threadIdx.x) >> 5;  // 0..511
    int d = 1 + (wid >> 4);          // 1..32
    int n0 = (wid & 15) << 4;
    const _Float16* Bm = cw16 + (size_t)d * THH * TIN + (size_t)n0 * TIN;
    v8f acc0 = {}, acc1 = {}, acc2 = {}, acc3 = {};
#pragma unroll
    for (int kc = 0; kc < THH; kc += 32) {
        v16h bf = frag_b_f16(Bm, TIN, kc);
        acc0 = wmma_acc(frag_a_f16(h16 +  0 * THH, THH, kc), bf, acc0);
        acc1 = wmma_acc(frag_a_f16(h16 + 16 * THH, THH, kc), bf, acc1);
        acc2 = wmma_acc(frag_a_f16(h16 + 32 * THH, THH, kc), bf, acc2);
        acc3 = wmma_acc(frag_a_f16(h16 + 48 * THH, THH, kc), bf, acc3);
    }
    int dst = ptr + d; if (dst >= TD) dst -= TD;
    float* out = buf + (size_t)dst * TB * THH;
    int lane = threadIdx.x & 31;
    int n = n0 + (lane & 15);
    int mo = 8 * (lane >> 4);
#pragma unroll
    for (int r = 0; r < 8; ++r) {
        out[( 0 + r + mo) * THH + n] += acc0[r];
        out[(16 + r + mo) * THH + n] += acc1[r];
        out[(32 + r + mo) * THH + n] += acc2[r];
        out[(48 + r + mo) * THH + n] += acc3[r];
    }
    if (blockIdx.x == 0) {  // zero consumed row (no wave writes row ptr: d>=1)
        float* zr = buf + (size_t)ptr * TB * THH;
        for (int i = threadIdx.x; i < TB * THH; i += blockDim.x) zr[i] = 0.0f;
    }
}

// snapshot: at t == lengths[b]-1, dbuf[b,d,h] = buf[d,b,h]; ptr[b] = (t+1)%33
__global__ void k_enc_snapshot(const float* __restrict__ buf, const int* __restrict__ lengths,
                               int t, float* __restrict__ dbuf, int* __restrict__ ptr) {
    int b = blockIdx.x;
    if (lengths[b] - 1 != t) return;
    for (int i = threadIdx.x; i < DHSZ; i += blockDim.x) {
        int d = i >> 8, h = i & 255;
        dbuf[(size_t)b * DHSZ + i] = buf[((size_t)d * TB + b) * THH + h];
    }
    if (threadIdx.x == 0) { int np = t + 1; ptr[b] = np % TD; }
}

// ---------------- decoder (three launches per iteration) ----------------

// waves 0..63:  h16 = f16(tanh(cur @ w_aff^T + b_aff + dbuf[b,ptr[b]]))
// waves 64..127: y = dbuf[b,ptr[b]] @ w_eff^T + b_eff
__global__ void k_dec_hy(const _Float16* __restrict__ cur16,
                         const _Float16* __restrict__ w_aff16, const float* __restrict__ b_aff,
                         const _Float16* __restrict__ w_eff16, const float* __restrict__ b_eff,
                         const float* __restrict__ dbuf, const int* __restrict__ ptr,
                         _Float16* __restrict__ h16, float* __restrict__ y) {
    int wid = (blockIdx.x * blockDim.x + threadIdx.x) >> 5;  // 0..127
    int lane = threadIdx.x & 31;
    if (wid < 64) {
        int m0 = (wid >> 4) << 4, n0 = (wid & 15) << 4;
        const _Float16* Bm = w_aff16 + (size_t)n0 * TIN;
        v8f acc = {};
#pragma unroll
        for (int kc = 0; kc < TIN; kc += 32)
            acc = wmma_acc(frag_a_f16(cur16 + (size_t)m0 * TIN, TIN, kc),
                           frag_b_f16(Bm, TIN, kc), acc);
        int n = n0 + (lane & 15);
#pragma unroll
        for (int r = 0; r < 8; ++r) {
            int m = m0 + r + 8 * (lane >> 4);
            float hd = dbuf[(size_t)m * DHSZ + (size_t)ptr[m] * THH + n];
            h16[m * THH + n] = (_Float16)tanhf(acc[r] + b_aff[n] + hd);
        }
    } else {
        int tile = wid - 64;
        int m0 = (tile >> 4) << 4, n0 = (tile & 15) << 4;
        const _Float16* Bm = w_eff16 + (size_t)n0 * THH;
        v8f acc = {};
#pragma unroll
        for (int kc = 0; kc < THH; kc += 32)
            acc = wmma_acc(frag_a_gather(dbuf, ptr, m0, kc),
                           frag_b_f16(Bm, THH, kc), acc);
        int n = n0 + (lane & 15);
#pragma unroll
        for (int r = 0; r < 8; ++r) {
            int m = m0 + r + 8 * (lane >> 4);
            y[m * TV + n] = acc[r] + b_eff[n];
        }
    }
}

// dbuf[b,(ptr[b]+d)%33] += h @ cw[d]^T (masked by done in think phase); zero row ptr[b]
// Same 64x16 column-strip decomposition as k_enc_update.
__global__ void k_dec_update(const _Float16* __restrict__ h16,
                             const _Float16* __restrict__ cw16,
                             float* __restrict__ dbuf, const int* __restrict__ ptr,
                             const int* __restrict__ done, int phase) {
    int wid = (blockIdx.x * blockDim.x + threadIdx.x) >> 5;  // 0..511
    int d = 1 + (wid >> 4);
    int n0 = (wid & 15) << 4;
    const _Float16* Bm = cw16 + (size_t)d * THH * TIN + (size_t)n0 * TIN;
    v8f acc0 = {}, acc1 = {}, acc2 = {}, acc3 = {};
#pragma unroll
    for (int kc = 0; kc < THH; kc += 32) {
        v16h bf = frag_b_f16(Bm, TIN, kc);
        acc0 = wmma_acc(frag_a_f16(h16 +  0 * THH, THH, kc), bf, acc0);
        acc1 = wmma_acc(frag_a_f16(h16 + 16 * THH, THH, kc), bf, acc1);
        acc2 = wmma_acc(frag_a_f16(h16 + 32 * THH, THH, kc), bf, acc2);
        acc3 = wmma_acc(frag_a_f16(h16 + 48 * THH, THH, kc), bf, acc3);
    }
    int lane = threadIdx.x & 31;
    int n = n0 + (lane & 15);
    int mo = 8 * (lane >> 4);
#pragma unroll
    for (int s = 0; s < 4; ++s) {
        v8f acc = (s == 0) ? acc0 : (s == 1) ? acc1 : (s == 2) ? acc2 : acc3;
#pragma unroll
        for (int r = 0; r < 8; ++r) {
            int m = s * 16 + r + mo;
            if (phase || !done[m]) {
                int dst = ptr[m] + d; if (dst >= TD) dst -= TD;
                dbuf[(size_t)m * DHSZ + dst * THH + n] += acc[r];
            }
        }
    }
    if (blockIdx.x == 0) {  // zero consumed rows (d>=1 never targets row ptr[m])
        for (int i = threadIdx.x; i < TB * THH; i += blockDim.x) {
            int m = i >> 8;
            if (phase || !done[m])
                dbuf[(size_t)m * DHSZ + (size_t)ptr[m] * THH + (i & 255)] = 0.0f;
        }
    }
}

// Gumbel-argmax sample; ST-estimator forward value is exactly the one-hot.
__global__ void k_dec_sample(const float* __restrict__ y, _Float16* __restrict__ cur16,
                             int* __restrict__ ptr, int* __restrict__ done,
                             int* __restrict__ steps, int phase, int key, int kk,
                             float* __restrict__ out) {
    __shared__ float sv[TV];
    __shared__ int   si[TV];
    int b = blockIdx.x, c = threadIdx.x;
    float yv = y[b * TV + c];
    sv[c] = yv + gumbelf((uint32_t)b, (uint32_t)key, (uint32_t)c);
    si[c] = c;
    __syncthreads();
#pragma unroll
    for (int s = 128; s > 0; s >>= 1) {
        if (c < s) {
            float v1 = sv[c], v2 = sv[c + s];
            int i1 = si[c], i2 = si[c + s];
            if (v2 > v1 || (v2 == v1 && i2 < i1)) { sv[c] = v2; si[c] = i2; }
        }
        __syncthreads();
    }
    int idx = si[0];
    if (phase == 0) {  // think: done-masked state updates
        int d0 = done[b];
        int nst = steps[b] + (d0 ? 0 : 1);
        int nd = d0 || (idx == TV - 1) || (nst > 8);
        if (!d0) cur16[b * TV + c] = (_Float16)((c == idx) ? 1.0f : 0.0f);
        if (c == 0) {
            steps[b] = nst;
            done[b] = nd;
            if (!d0) { int p = ptr[b] + 1; ptr[b] = (p >= TD) ? 0 : p; }
        }
    } else {           // decode: unconditional; emit logits (drop last class)
        if (c < TV - 1) out[((size_t)b * TN + kk) * (TV - 1) + c] = yv;
        cur16[b * TV + c] = (_Float16)((c == idx) ? 1.0f : 0.0f);
        if (c == 0) { int p = ptr[b] + 1; ptr[b] = (p >= TD) ? 0 : p; }
    }
}

__global__ void k_write_steps(const int* __restrict__ steps, float* __restrict__ out) {
    int b = threadIdx.x;
    if (b < TB) out[b] = (float)steps[b];
}

// ---------------- host ----------------

static inline size_t alup(size_t x) { return (x + 255) & ~(size_t)255; }

extern "C" void kernel_launch(void* const* d_in, const int* in_sizes, int n_in,
                              void* d_out, int out_size, void* d_ws, size_t ws_size,
                              hipStream_t stream) {
    (void)in_sizes; (void)n_in; (void)out_size; (void)ws_size;
    const float* x       = (const float*)d_in[0];
    const float* w_aff   = (const float*)d_in[1];
    const float* b_aff   = (const float*)d_in[2];
    const float* lateral = (const float*)d_in[3];
    const float* tau     = (const float*)d_in[4];
    const float* w_eff   = (const float*)d_in[5];
    const float* b_eff   = (const float*)d_in[6];
    const int*   lengths = (const int*)d_in[7];
    // d_in[8] (N) is a device scalar; N = T-3 = 125 is fixed by the reference.
    float* out = (float*)d_out;

    // workspace carve
    char* w = (char*)d_ws;
    size_t off = 0;
    _Float16* cw16    = (_Float16*)(w + off); off += alup((size_t)TD * THH * TIN * 2);
    _Float16* w_aff16 = (_Float16*)(w + off); off += alup((size_t)THH * TIN * 2);
    _Float16* w_eff16 = (_Float16*)(w + off); off += alup((size_t)TV * THH * 2);
    float*    buf     = (float*)   (w + off); off += alup((size_t)TD * TB * THH * 4);
    float*    dbuf    = (float*)   (w + off); off += alup((size_t)TB * DHSZ * 4);
    _Float16* h16     = (_Float16*)(w + off); off += alup((size_t)TB * THH * 2);
    _Float16* cur16   = (_Float16*)(w + off); off += alup((size_t)TB * TIN * 2);
    float*    y       = (float*)   (w + off); off += alup((size_t)TB * TV * 4);
    int*      ptrA    = (int*)     (w + off); off += alup(TB * 4);
    int*      done    = (int*)     (w + off); off += alup(TB * 4);
    int*      steps   = (int*)     (w + off); off += alup(TB * 4);

    // prep (re-done every call: deterministic, no cross-call state)
    k_cvt_f16<<<(THH * TIN + 255) / 256, 256, 0, stream>>>(w_aff, w_aff16, THH * TIN);
    k_cvt_f16<<<(TV * THH + 255) / 256, 256, 0, stream>>>(w_eff, w_eff16, TV * THH);
    k_prep_cw<<<(TD * THH * TIN) / 256, 256, 0, stream>>>(tau, lateral, cw16);
    k_zero_f32<<<(TD * TB * THH) / 256, 256, 0, stream>>>(buf, TD * TB * THH);
    k_init_dec<<<1, 64, 0, stream>>>(done, steps);
    k_gather_decin<<<TB, TIN, 0, stream>>>(x, lengths, cur16);

    // encoder: 128 sequential steps
    for (int t = 0; t < TT; ++t) {
        int ptr = t % TD;
        k_enc_h<<<16, 128, 0, stream>>>(x, t, w_aff16, b_aff, buf, ptr, h16);
        k_enc_update<<<128, 128, 0, stream>>>(h16, cw16, buf, ptr);
        k_enc_snapshot<<<TB, 256, 0, stream>>>(buf, lengths, t, dbuf, ptrA);
    }

    // think phase: MAX_THINK+1 = 9 masked iterations
    for (int k = 0; k < 9; ++k) {
        k_dec_hy<<<32, 128, 0, stream>>>(cur16, w_aff16, b_aff, w_eff16, b_eff,
                                         dbuf, ptrA, h16, y);
        k_dec_update<<<128, 128, 0, stream>>>(h16, cw16, dbuf, ptrA, done, 0);
        k_dec_sample<<<TB, TV, 0, stream>>>(y, cur16, ptrA, done, steps, 0, k, 0, out);
    }
    k_write_steps<<<1, 64, 0, stream>>>(steps, out + (size_t)TB * TN * (TV - 1));

    // decode phase: N = 125 unconditional iterations, logits streamed to d_out
    for (int k = 0; k < TN; ++k) {
        k_dec_hy<<<32, 128, 0, stream>>>(cur16, w_aff16, b_aff, w_eff16, b_eff,
                                         dbuf, ptrA, h16, y);
        k_dec_update<<<128, 128, 0, stream>>>(h16, cw16, dbuf, ptrA, done, 1);
        k_dec_sample<<<TB, TV, 0, stream>>>(y, cur16, ptrA, done, steps, 1, 1000 + k, k, out);
    }
}
